// C_VAE_TF2_23012434772053
// MI455X (gfx1250) — matrix operand order repaired and checked by reading, not compiled, not run on hardware
//
#include <hip/hip_runtime.h>
#include <hip/hip_bf16.h>

// ---------------------------------------------------------------------------
// C-VAE LSTM for MI455X (gfx1250): bf16 WMMA, fused gate+cell quad-tiles,
// LDS-staged weight panels, persistent grid kernel with device barriers.
// ---------------------------------------------------------------------------

typedef __attribute__((ext_vector_type(16))) __bf16 bf16x16;
typedef __attribute__((ext_vector_type(8)))  float  floatx8;

constexpr int B_  = 128;
constexpr int T_  = 256;
constexpr int D_  = 128;
constexpr int Z_  = 64;
constexpr int E_  = 512;
constexpr int DC_ = 512;
constexpr int H_  = 512;    // gate width (E_ == DC_)
constexpr int NG  = 2048;   // 4*H
constexpr int NBLK = 32;    // persistent workgroups == column groups of H
constexpr int NTHR = 256;   // 8 wave32 waves per workgroup == m tiles
constexpr int KC   = 128;   // K-chunk staged in LDS
constexpr int KCP  = KC + 8; // padded LDS stride (bank-conflict free)

struct RParams {
  const __bf16 *WsumT, *WbotT, *e0UT, *e1WT, *e1UT, *msWT;
  const __bf16 *d0WT, *d0UT, *d1WT, *d1UT, *oWT, *xbf;
  __bf16 *eh0, *eh1, *dh0, *dh1, *m, *zbf;
  float  *ec0, *ec1, *dc0, *dc1, *biascat;
  const float *enc0_b, *enc1_b, *dec0_b, *dec1_b, *out_b, *eps;
  float *decoded, *o_sig, *o_mu, *o_ls, *o_z;
  unsigned *bar_cnt, *bar_gen;
};

#define DEVFN static __device__ __forceinline__

DEVFN float sigm(float x) { return 1.0f / (1.0f + __expf(-x)); }

union ABFrag { uint4 u[2]; bf16x16 v; };

// A fragment: row-major [M,K] bf16. lanes0-15 K=k..k+15, lanes16-31
// K=k+16..k+31.
DEVFN bf16x16 load_a(const __bf16* A, int lda, int row, int k, int lane) {
  ABFrag t;
  const __bf16* p = A + (size_t)row * lda + k + ((lane >> 4) << 4);
  t.u[0] = *(const uint4*)(p);
  t.u[1] = *(const uint4*)(p + 8);
  return t.v;
}

// B fragment from [N,K]-transposed panel: lanes0-15 K=k..k+15 of column n,
// lanes16-31 K=k+16..k+31.
DEVFN bf16x16 load_b(const __bf16* BT, int ldb, int col, int k, int lane) {
  ABFrag t;
  const __bf16* p = BT + (size_t)col * ldb + k + ((lane >> 4) << 4);
  t.u[0] = *(const uint4*)(p);
  t.u[1] = *(const uint4*)(p + 8);
  return t.v;
}

// Same fragment shape but sourced from an LDS-staged panel (stride KCP).
DEVFN bf16x16 load_b_lds(const __bf16* base, int lane, int k) {
  ABFrag t;
  const __bf16* p = base + (size_t)(lane & 15) * KCP + k + ((lane >> 4) << 4);
  t.u[0] = *(const uint4*)(p);
  t.u[1] = *(const uint4*)(p + 8);
  return t.v;
}

struct GemmIn { const __bf16* A; int lda; const __bf16* BT; int K; };

DEVFN void grid_barrier(unsigned* cnt, unsigned* gen) {
  __syncthreads();
  if (threadIdx.x == 0) {
    __threadfence();
    unsigned g = __hip_atomic_load(gen, __ATOMIC_RELAXED, __HIP_MEMORY_SCOPE_AGENT);
    if (__hip_atomic_fetch_add(cnt, 1u, __ATOMIC_ACQ_REL, __HIP_MEMORY_SCOPE_AGENT)
        == (unsigned)NBLK - 1u) {
      __hip_atomic_store(cnt, 0u, __ATOMIC_RELAXED, __HIP_MEMORY_SCOPE_AGENT);
      __hip_atomic_fetch_add(gen, 1u, __ATOMIC_ACQ_REL, __HIP_MEMORY_SCOPE_AGENT);
    } else {
      while (__hip_atomic_load(gen, __ATOMIC_ACQUIRE, __HIP_MEMORY_SCOPE_AGENT) == g)
        __builtin_amdgcn_s_sleep(2);
    }
  }
  __syncthreads();
}

// Fused gate-GEMM + LSTM cell. Block handles column group nh (16 cols of H),
// wave wv handles m-tile wv. For each K-chunk the block stages the four gate
// weight panels into LDS; each wave then does 4 WMMAs per k-step sharing one
// A fragment. Cell update (sigmoid/tanh, c/h) happens in-register.
template <int NIN>
DEVFN void gate_phase(const GemmIn* in, const float* __restrict__ bias,
                      float* __restrict__ cSt, __bf16* __restrict__ hSt,
                      __bf16* lsB, int nh, int wv, int lane) {
  floatx8 acc[4] = {{}, {}, {}, {}};
  const int row = (wv << 4) + (lane & 15);

#pragma unroll
  for (int j = 0; j < NIN; ++j) {
    const __bf16* A  = in[j].A;
    const __bf16* BT = in[j].BT;
    const int lda = in[j].lda;
    const int K   = in[j].K;
    for (int kc = 0; kc < K; kc += KC) {
      const int klen = (K - kc < KC) ? (K - kc) : KC;
      __syncthreads();   // previous chunk fully consumed
      // Cooperative stage: 4 gates x 16 cols x KC k-values (b128 copies).
      for (int i = threadIdx.x; i < (4 * 16 * KC) / 8; i += NTHR) {
        const int e  = i << 3;
        const int kk = e & (KC - 1);
        const int c  = (e >> 7) & 15;
        const int g  = e >> 11;
        const __bf16* src = BT + (size_t)(g * H_ + (nh << 4) + c) * K + kc + kk;
        if (kc + KC < K) __builtin_prefetch(src + KC, 0, 3);
        *(uint4*)(lsB + (size_t)((g << 4) + c) * KCP + kk) = *(const uint4*)src;
      }
      __syncthreads();   // panel visible to all waves
      for (int k4 = 0; k4 < klen; k4 += 32) {
        bf16x16 a = load_a(A, lda, row, kc + k4, lane);
#pragma unroll
        for (int g = 0; g < 4; ++g) {
          bf16x16 b = load_b_lds(lsB + (size_t)(g << 4) * KCP, lane, k4);
          acc[g] = __builtin_amdgcn_wmma_f32_16x16x32_bf16(
              false, a, false, b, (short)0, acc[g], false, false);
        }
      }
    }
  }

  // In-register LSTM cell: gates [i, f, cg, o], each H_ wide.
  const int colH = (nh << 4) + (lane & 15);
  const float bi = bias[colH];
  const float bf = bias[colH + H_];
  const float bc = bias[colH + 2 * H_];
  const float bo = bias[colH + 3 * H_];
  const int rbase = (wv << 4) + ((lane >> 4) << 3);
#pragma unroll
  for (int r = 0; r < 8; ++r) {
    const size_t idx = (size_t)(rbase + r) * H_ + colH;
    float iv = sigm(acc[0][r] + bi);
    float fv = sigm(acc[1][r] + bf);
    float cg = tanhf(acc[2][r] + bc);
    float ov = sigm(acc[3][r] + bo);
    float c2 = fv * cSt[idx] + iv * cg;
    cSt[idx] = c2;
    hSt[idx] = (__bf16)(ov * tanhf(c2));
  }
}

// Fused [mu | log_sigma] GEMM + reparameterization. Pair-tile: one wave owns
// mu cols nz*16.. and ls cols 64+nz*16.. sharing the A fragment.
DEVFN void musigz_phase(int t, const RParams& p, int gw, int lane) {
  if (gw >= 32) return;
  const int mt = gw & 7, nz = gw >> 3;
  const int row = (mt << 4) + (lane & 15);
  const int c0  = (nz << 4) + (lane & 15);
  floatx8 amu = {}, als = {};
  for (int k = 0; k < E_; k += 32) {
    bf16x16 a  = load_a(p.eh1, E_, row, k, lane);
    bf16x16 bm = load_b(p.msWT, E_, c0, k, lane);
    bf16x16 bs = load_b(p.msWT, E_, c0 + Z_, k, lane);
    amu = __builtin_amdgcn_wmma_f32_16x16x32_bf16(false, a, false, bm, (short)0,
                                                  amu, false, false);
    als = __builtin_amdgcn_wmma_f32_16x16x32_bf16(false, a, false, bs, (short)0,
                                                  als, false, false);
  }
  const float bmu = p.biascat[c0];
  const float bls = p.biascat[c0 + Z_];
  const int rbase = (mt << 4) + ((lane >> 4) << 3);
#pragma unroll
  for (int r = 0; r < 8; ++r) {
    const int rw = rbase + r;
    const size_t o = (size_t)t * B_ * Z_ + (size_t)rw * Z_ + c0;
    float mu = amu[r] + bmu;
    float ls = als[r] + bls;
    float sg = __expf(ls);
    float z  = mu + sg * p.eps[o];
    p.o_mu[o]  = mu;
    p.o_ls[o]  = ls;
    p.o_sig[o] = sg;
    p.o_z[o]   = z;
    p.zbf[rw * Z_ + c0] = (__bf16)z;
  }
}

// Output dense + sigmoid; writes decoded[b,t,:] and m = -sigmoid(c_t).
DEVFN void out_phase(int t, const RParams& p, int gw, int lane) {
  if (gw >= 64) return;
  const int mt = gw & 7, nd = gw >> 3;   // 8 x 8 tiles over [128,128]
  const int row = (mt << 4) + (lane & 15);
  const int col = (nd << 4) + (lane & 15);
  floatx8 acc = {};
  for (int k = 0; k < DC_; k += 32) {
    bf16x16 a = load_a(p.dh1, DC_, row, k, lane);
    bf16x16 b = load_b(p.oWT, DC_, col, k, lane);
    acc = __builtin_amdgcn_wmma_f32_16x16x32_bf16(false, a, false, b, (short)0,
                                                  acc, false, false);
  }
  const float bv = p.out_b[col];
  const int rbase = (mt << 4) + ((lane >> 4) << 3);
#pragma unroll
  for (int r = 0; r < 8; ++r) {
    const int b = rbase + r;
    float ct = sigm(acc[r] + bv);
    p.decoded[(size_t)b * T_ * D_ + (size_t)t * D_ + col] = ct;
    p.m[b * D_ + col] = (__bf16)(-sigm(ct));  // -sigmoid(c_prev) for next step
  }
}

// ----------------------------- kernels -------------------------------------

__global__ void k_transpose_bf16(const float* __restrict__ src,
                                 __bf16* __restrict__ dst, int K, int N) {
  const size_t total = (size_t)K * N;
  const size_t str = (size_t)gridDim.x * blockDim.x;
  for (size_t i = (size_t)blockIdx.x * blockDim.x + threadIdx.x; i < total; i += str) {
    const size_t n = i / K, k = i - n * K;
    dst[i] = (__bf16)src[k * (size_t)N + n];
  }
}

__global__ void k_enc0_prep(const float* __restrict__ W,
                            __bf16* __restrict__ WsumT,
                            __bf16* __restrict__ WbotT) {
  const size_t total = (size_t)NG * D_;
  const size_t str = (size_t)gridDim.x * blockDim.x;
  for (size_t i = (size_t)blockIdx.x * blockDim.x + threadIdx.x; i < total; i += str) {
    const size_t n = i / D_, k = i - n * D_;
    float top = W[k * NG + n];
    float bot = W[(k + D_) * NG + n];
    WsumT[i] = (__bf16)(top + bot);
    WbotT[i] = (__bf16)bot;
  }
}

__global__ void k_musig_prep(const float* __restrict__ muW,
                             const float* __restrict__ sigW,
                             const float* __restrict__ mub,
                             const float* __restrict__ sigb,
                             __bf16* __restrict__ msWT, float* __restrict__ biascat) {
  const size_t total = (size_t)(2 * Z_) * E_;
  const size_t str = (size_t)gridDim.x * blockDim.x;
  const size_t tid0 = (size_t)blockIdx.x * blockDim.x + threadIdx.x;
  for (size_t i = tid0; i < total; i += str) {
    const size_t n = i / E_, k = i - n * E_;
    float v = (n < Z_) ? muW[k * Z_ + n] : sigW[k * Z_ + (n - Z_)];
    msWT[i] = (__bf16)v;
  }
  for (size_t j = tid0; j < (size_t)(2 * Z_); j += str)
    biascat[j] = (j < Z_) ? mub[j] : sigb[j - Z_];
}

__global__ void k_xconv(const float* __restrict__ src, __bf16* __restrict__ dst,
                        size_t total) {
  const size_t str = (size_t)gridDim.x * blockDim.x;
  for (size_t i = (size_t)blockIdx.x * blockDim.x + threadIdx.x; i < total; i += str)
    dst[i] = (__bf16)src[i];
}

__global__ void k_init(RParams p) {
  const int tid = blockIdx.x * blockDim.x + threadIdx.x;
  const int tc = gridDim.x * blockDim.x;
  for (int i = tid; i < B_ * E_; i += tc) {
    p.eh0[i] = (__bf16)0.0f; p.eh1[i] = (__bf16)0.0f;
    p.dh0[i] = (__bf16)0.0f; p.dh1[i] = (__bf16)0.0f;
    p.ec0[i] = 0.0f; p.ec1[i] = 0.0f; p.dc0[i] = 0.0f; p.dc1[i] = 0.0f;
  }
  for (int i = tid; i < B_ * D_; i += tc) p.m[i] = (__bf16)(-0.5f);  // -sigmoid(0)
  for (int i = tid; i < B_ * Z_; i += tc) p.zbf[i] = (__bf16)0.0f;
  if (tid == 0) { *p.bar_cnt = 0u; *p.bar_gen = 0u; }
}

__global__ void __launch_bounds__(NTHR, 1) k_recurrent(RParams p) {
  __shared__ __bf16 lsB[4 * 16 * KCP];
  const int lane = threadIdx.x & 31;
  const int wv   = threadIdx.x >> 5;       // m-tile (0..7)
  const int nh   = blockIdx.x;             // H column group (0..31)
  const int gw   = blockIdx.x * (NTHR / 32) + wv;

  for (int t = 0; t < T_; ++t) {
    // enc0: x_t@(Wtop+Wbot) + m@Wbot + eh0@U + b   (m = -sigmoid(c_prev))
    {
      GemmIn in[3] = { { p.xbf + (size_t)t * D_, T_ * D_, p.WsumT, D_ },
                       { p.m,   D_, p.WbotT, D_ },
                       { p.eh0, E_, p.e0UT,  E_ } };
      gate_phase<3>(in, p.enc0_b, p.ec0, p.eh0, lsB, nh, wv, lane);
    }
    grid_barrier(p.bar_cnt, p.bar_gen);

    // enc1
    {
      GemmIn in[2] = { { p.eh0, E_, p.e1WT, E_ },
                       { p.eh1, E_, p.e1UT, E_ } };
      gate_phase<2>(in, p.enc1_b, p.ec1, p.eh1, lsB, nh, wv, lane);
    }
    grid_barrier(p.bar_cnt, p.bar_gen);

    // [mu | log_sigma] + reparameterize
    musigz_phase(t, p, gw, lane);
    grid_barrier(p.bar_cnt, p.bar_gen);

    // dec0
    {
      GemmIn in[2] = { { p.zbf, Z_,  p.d0WT, Z_  },
                       { p.dh0, DC_, p.d0UT, DC_ } };
      gate_phase<2>(in, p.dec0_b, p.dc0, p.dh0, lsB, nh, wv, lane);
    }
    grid_barrier(p.bar_cnt, p.bar_gen);

    // dec1
    {
      GemmIn in[2] = { { p.dh0, DC_, p.d1WT, DC_ },
                       { p.dh1, DC_, p.d1UT, DC_ } };
      gate_phase<2>(in, p.dec1_b, p.dc1, p.dh1, lsB, nh, wv, lane);
    }
    grid_barrier(p.bar_cnt, p.bar_gen);

    // output dense + sigmoid
    out_phase(t, p, gw, lane);
    grid_barrier(p.bar_cnt, p.bar_gen);
  }
}

// ----------------------------- launcher ------------------------------------

extern "C" void kernel_launch(void* const* d_in, const int* in_sizes, int n_in,
                              void* d_out, int out_size, void* d_ws, size_t ws_size,
                              hipStream_t stream) {
  (void)in_sizes; (void)n_in; (void)out_size; (void)ws_size;
  const float* x      = (const float*)d_in[0];
  const float* eps    = (const float*)d_in[1];
  const float* enc0_W = (const float*)d_in[2];
  const float* enc0_U = (const float*)d_in[3];
  const float* enc0_b = (const float*)d_in[4];
  const float* enc1_W = (const float*)d_in[5];
  const float* enc1_U = (const float*)d_in[6];
  const float* enc1_b = (const float*)d_in[7];
  const float* mu_W   = (const float*)d_in[8];
  const float* mu_b   = (const float*)d_in[9];
  const float* sig_W  = (const float*)d_in[10];
  const float* sig_b  = (const float*)d_in[11];
  const float* dec0_W = (const float*)d_in[12];
  const float* dec0_U = (const float*)d_in[13];
  const float* dec0_b = (const float*)d_in[14];
  const float* dec1_W = (const float*)d_in[15];
  const float* dec1_U = (const float*)d_in[16];
  const float* dec1_b = (const float*)d_in[17];
  const float* out_W  = (const float*)d_in[18];
  const float* out_b  = (const float*)d_in[19];

  char* ws = (char*)d_ws;
  size_t off = 0;
  auto take = [&](size_t bytes) -> char* {
    off = (off + 255) & ~(size_t)255;
    char* p = ws + off;
    off += bytes;
    return p;
  };

  RParams P;
  P.bar_cnt = (unsigned*)take(sizeof(unsigned));
  P.bar_gen = (unsigned*)take(sizeof(unsigned));

  __bf16* WsumT = (__bf16*)take((size_t)NG * D_ * 2);
  __bf16* WbotT = (__bf16*)take((size_t)NG * D_ * 2);
  __bf16* e0UT  = (__bf16*)take((size_t)NG * E_ * 2);
  __bf16* e1WT  = (__bf16*)take((size_t)NG * E_ * 2);
  __bf16* e1UT  = (__bf16*)take((size_t)NG * E_ * 2);
  __bf16* msWT  = (__bf16*)take((size_t)(2 * Z_) * E_ * 2);
  __bf16* d0WT  = (__bf16*)take((size_t)NG * Z_ * 2 + 512);  // +pad: chunk overrun
  __bf16* d0UT  = (__bf16*)take((size_t)NG * DC_ * 2);
  __bf16* d1WT  = (__bf16*)take((size_t)NG * DC_ * 2);
  __bf16* d1UT  = (__bf16*)take((size_t)NG * DC_ * 2);
  __bf16* oWT   = (__bf16*)take((size_t)D_ * DC_ * 2);
  __bf16* xbf   = (__bf16*)take((size_t)B_ * T_ * D_ * 2);

  P.eh0 = (__bf16*)take((size_t)B_ * E_ * 2);
  P.eh1 = (__bf16*)take((size_t)B_ * E_ * 2);
  P.dh0 = (__bf16*)take((size_t)B_ * DC_ * 2);
  P.dh1 = (__bf16*)take((size_t)B_ * DC_ * 2);
  P.m   = (__bf16*)take((size_t)B_ * D_ * 2);
  P.zbf = (__bf16*)take((size_t)B_ * Z_ * 2);

  P.ec0 = (float*)take((size_t)B_ * E_ * 4);
  P.ec1 = (float*)take((size_t)B_ * E_ * 4);
  P.dc0 = (float*)take((size_t)B_ * DC_ * 4);
  P.dc1 = (float*)take((size_t)B_ * DC_ * 4);
  P.biascat = (float*)take((size_t)(2 * Z_) * 4);

  P.WsumT = WsumT; P.WbotT = WbotT; P.e0UT = e0UT; P.e1WT = e1WT; P.e1UT = e1UT;
  P.msWT = msWT; P.d0WT = d0WT; P.d0UT = d0UT; P.d1WT = d1WT; P.d1UT = d1UT;
  P.oWT = oWT; P.xbf = xbf;
  P.enc0_b = enc0_b; P.enc1_b = enc1_b; P.dec0_b = dec0_b; P.dec1_b = dec1_b;
  P.out_b = out_b; P.eps = eps;

  float* out = (float*)d_out;
  P.decoded = out;
  P.o_sig = out + (size_t)B_ * T_ * D_;
  P.o_mu  = P.o_sig + (size_t)T_ * B_ * Z_;
  P.o_ls  = P.o_mu  + (size_t)T_ * B_ * Z_;
  P.o_z   = P.o_ls  + (size_t)T_ * B_ * Z_;

  // weight prep (parallel, once per call)
  k_enc0_prep<<<512, 256, 0, stream>>>(enc0_W, WsumT, WbotT);
  k_transpose_bf16<<<1024, 256, 0, stream>>>(enc0_U, e0UT, E_, NG);
  k_transpose_bf16<<<1024, 256, 0, stream>>>(enc1_W, e1WT, E_, NG);
  k_transpose_bf16<<<1024, 256, 0, stream>>>(enc1_U, e1UT, E_, NG);
  k_musig_prep<<<256, 256, 0, stream>>>(mu_W, sig_W, mu_b, sig_b, msWT, P.biascat);
  k_transpose_bf16<<<512, 256, 0, stream>>>(dec0_W, d0WT, Z_, NG);
  k_transpose_bf16<<<1024, 256, 0, stream>>>(dec0_U, d0UT, DC_, NG);
  k_transpose_bf16<<<1024, 256, 0, stream>>>(dec1_W, d1WT, DC_, NG);
  k_transpose_bf16<<<1024, 256, 0, stream>>>(dec1_U, d1UT, DC_, NG);
  k_transpose_bf16<<<256, 256, 0, stream>>>(out_W, oWT, DC_, D_);
  k_xconv<<<2048, 256, 0, stream>>>(x, xbf, (size_t)B_ * T_ * D_);
  k_init<<<256, 256, 0, stream>>>(P);

  // persistent recurrent kernel over all T steps
  k_recurrent<<<NBLK, NTHR, 0, stream>>>(P);
}
